// MBConvApprox_68289980006919
// MI455X (gfx1250) — matrix operand order repaired and checked by reading, not compiled, not run on hardware
//
#include <hip/hip_runtime.h>
#include <stdint.h>

typedef __attribute__((ext_vector_type(16))) __bf16 v16bf;
typedef __attribute__((ext_vector_type(8)))  float  v8f;

#define C_IN   96
#define M_EXP  576
#define HW_    56
#define EPS_   1e-5f

// spatial tile: 4x14 interior, 6x16 halo
#define TR_    4
#define TC_    14
#define PHALO  96     // 6*16 halo pixels
#define PPAD   64     // interior pixels padded to 4 WMMA n-tiles
#define PINT   56     // 4*14 interior pixels

// workspace layout (bytes)
#define WS_W1B 0              // 576*96  bf16 = 110592
#define WS_W3B 110592         // 96*576  bf16 = 110592
#define WS_B1  221184         // 576 f32
#define WS_B2  223488         // 576 f32
#define WS_B3  225792         // 96  f32
#define WS_WD  226176         // 576*9 f32 = 20736 -> ends 246912
#define WS_XP  247040         // 256B-aligned; packed bf16 x: 32*48*56*56*4 B
#define XP_ELEMS (32u * 48u * (HW_ * HW_))
#define WS_NEED  (WS_XP + (size_t)XP_ELEMS * 4u)

// ---------------------------------------------------------------------------
__device__ __forceinline__ uint16_t f2bf(float f) {
  uint32_t u = __float_as_uint(f);
  uint32_t r = ((u >> 16) & 1u) + 0x7FFFu;   // round-to-nearest-even
  return (uint16_t)((u + r) >> 16);
}
__device__ __forceinline__ float bf2f(uint16_t h) {
  return __uint_as_float(((uint32_t)h) << 16);
}
// single-instruction v_cvt_pk_bf16_f32 when available
__device__ __forceinline__ uint32_t pack_bf16(float a, float b) {
#if __has_builtin(__builtin_amdgcn_cvt_pk_bf16_f32)
  auto t = __builtin_amdgcn_cvt_pk_bf16_f32(a, b);
  uint32_t u = 0;
  __builtin_memcpy(&u, &t, sizeof(t) < sizeof(u) ? sizeof(t) : sizeof(u));
  return u;
#else
  return (uint32_t)f2bf(a) | ((uint32_t)f2bf(b) << 16);
#endif
}

// gfx1250 async global->LDS copy (ASYNCcnt-tracked), guarded for portability
#if __has_builtin(__builtin_amdgcn_global_load_async_to_lds_b32)
#define HAS_ASYNC_LDS 1
__device__ __forceinline__ void async_ld32(const uint32_t* g, uint32_t* l) {
  __builtin_amdgcn_global_load_async_to_lds_b32(
      (__attribute__((address_space(1))) int*)(uintptr_t)g,
      (__attribute__((address_space(3))) int*)l, 0, 0);
}
__device__ __forceinline__ void wait_async0() {
#if __has_builtin(__builtin_amdgcn_s_wait_asynccnt)
  __builtin_amdgcn_s_wait_asynccnt(0);
#else
  asm volatile("s_wait_asynccnt 0x0" ::: "memory");
#endif
}
#else
#define HAS_ASYNC_LDS 0
#endif

// ---------------- prep: fold BN into weights/bias, convert to bf16 ----------
__global__ void mbconv_prep(const float* __restrict__ w1, const float* __restrict__ g1,
                            const float* __restrict__ b1, const float* __restrict__ m1,
                            const float* __restrict__ v1, const float* __restrict__ wd,
                            const float* __restrict__ g2, const float* __restrict__ b2,
                            const float* __restrict__ m2, const float* __restrict__ v2,
                            const float* __restrict__ w3, const float* __restrict__ g3,
                            const float* __restrict__ b3, const float* __restrict__ m3,
                            const float* __restrict__ v3,
                            uint16_t* __restrict__ w1b, uint16_t* __restrict__ w3b,
                            float* __restrict__ bias1, float* __restrict__ bias2,
                            float* __restrict__ bias3, float* __restrict__ wdf) {
  int i = blockIdx.x * blockDim.x + threadIdx.x;
  if (i < M_EXP * C_IN) {
    int o1 = i / C_IN;
    float s1 = g1[o1] * rsqrtf(v1[o1] + EPS_);
    w1b[i] = f2bf(w1[i] * s1);
    int o3 = i / M_EXP;
    float s3 = g3[o3] * rsqrtf(v3[o3] + EPS_);
    w3b[i] = f2bf(w3[i] * s3);
  }
  if (i < M_EXP * 9) {
    int m = i / 9;
    float s2 = g2[m] * rsqrtf(v2[m] + EPS_);
    wdf[i] = wd[i] * s2;
  }
  if (i < M_EXP) {
    float s1 = g1[i] * rsqrtf(v1[i] + EPS_);
    bias1[i] = b1[i] - m1[i] * s1;
    float s2 = g2[i] * rsqrtf(v2[i] + EPS_);
    bias2[i] = b2[i] - m2[i] * s2;
  }
  if (i < C_IN) {
    float s3 = g3[i] * rsqrtf(v3[i] + EPS_);
    bias3[i] = b3[i] - m3[i] * s3;
  }
}

// ---- prep 2: pre-pack x into bf16 channel-pair layout xp[nb][48][h][w] -----
__global__ void mbconv_xpack(const float* __restrict__ x, uint32_t* __restrict__ xp) {
  unsigned i = blockIdx.x * blockDim.x + threadIdx.x;
  if (i >= XP_ELEMS) return;
  unsigned pix = i % (HW_ * HW_);
  unsigned kp  = (i / (HW_ * HW_)) % 48;
  unsigned nb  = i / (HW_ * HW_ * 48);
  const float* px = x + ((size_t)nb * C_IN + 2 * kp) * (HW_ * HW_) + pix;
  xp[i] = pack_bf16(px[0], px[HW_ * HW_]);
}

// ---------------- fused MBConv tile kernel ----------------------------------
template <int USE_XP>
__global__ __launch_bounds__(256, 1)
void mbconv_fused(const float* __restrict__ x, const uint32_t* __restrict__ xp,
                  const uint16_t* __restrict__ w1b, const uint16_t* __restrict__ w3b,
                  const float* __restrict__ bias1, const float* __restrict__ bias2,
                  const float* __restrict__ bias3, const float* __restrict__ wdf,
                  float* __restrict__ out) {
  // t1p: expand output (bf16 channel pairs), [halo pixel][chan-pair]  110592 B
  // bpack: stage A = x tile [48][96]; stage C/D = depthwise out [288][64]  73728 B
  __shared__ uint32_t t1p[PHALO][M_EXP / 2];
  __shared__ uint32_t bpack[M_EXP / 2][PPAD];

  const int tid  = threadIdx.x;
  const int lane = tid & 31;
  const int wv   = tid >> 5;     // wave id 0..7
  const int n    = lane & 15;    // WMMA column / M-row lane index
  const int hi   = lane >> 4;    // lane half selects K/M sub-block

  const int tileid = blockIdx.x;       // 0..55
  const int nb     = blockIdx.y;       // batch
  const int row0   = (tileid >> 2) * TR_;
  const int col0   = (tileid & 3) * TC_;

  uint32_t (*xb)[PHALO] = (uint32_t(*)[PHALO])bpack;  // [48][96] packed bf16 pairs

  // ---- Stage A: x halo tile -> LDS (bf16 channel pairs, B-operand layout) --
  if (USE_XP) {
#if HAS_ASYNC_LDS
    // pre-zero (covers out-of-image halo), then async-copy valid pixels
    for (int idx = tid; idx < 48 * PHALO; idx += 256) xb[0][idx] = 0u;
    __syncthreads();
    for (int idx = tid; idx < 48 * PHALO; idx += 256) {
      int kp = idx / PHALO;
      int p  = idx - kp * PHALO;
      int h = row0 - 1 + (p >> 4);
      int w = col0 - 1 + (p & 15);
      if ((unsigned)h < HW_ && (unsigned)w < HW_) {
        const uint32_t* g = xp + (((size_t)nb * 48 + kp) * HW_ + h) * HW_ + w;
        async_ld32(g, &xb[kp][p]);
      }
    }
    wait_async0();
#else
    for (int idx = tid; idx < 48 * PHALO; idx += 256) {
      int kp = idx / PHALO;
      int p  = idx - kp * PHALO;
      int h = row0 - 1 + (p >> 4);
      int w = col0 - 1 + (p & 15);
      uint32_t val = 0u;
      if ((unsigned)h < HW_ && (unsigned)w < HW_)
        val = xp[(((size_t)nb * 48 + kp) * HW_ + h) * HW_ + w];
      xb[kp][p] = val;
    }
#endif
  } else {
    for (int idx = tid; idx < 48 * PHALO; idx += 256) {
      int kp = idx / PHALO;
      int p  = idx - kp * PHALO;
      int h = row0 - 1 + (p >> 4);
      int w = col0 - 1 + (p & 15);
      uint32_t val = 0u;
      if ((unsigned)h < HW_ && (unsigned)w < HW_) {
        const float* px = x + (((size_t)nb * C_IN + 2 * kp) * HW_ + h) * HW_ + w;
        val = pack_bf16(px[0], px[HW_ * HW_]);
      }
      xb[kp][p] = val;
    }
  }
  __syncthreads();

  // ---- Stage B: expand GEMM (576x96 @ 96xPHALO), BN1 + ReLU ---------------
  // 36 m-tiles x 6 pixel-tiles; each pixel-tile is one halo row (16 cols)
  for (int j = wv; j < 36 * 6; j += 8) {
    int mt = j / 6, pt = j - mt * 6;
    int h = row0 - 1 + pt;
    int w = col0 - 1 + n;
    bool valid = ((unsigned)h < HW_) && ((unsigned)w < HW_);
    v8f acc = {};
    const int arow = mt * 16 + n;
    for (int ks = 0; ks < 3; ++ks) {
      union { uint4 q[2]; v16bf v; } a;
      const uint16_t* ap = w1b + arow * C_IN + ks * 32 + hi * 16;
      a.q[0] = *(const uint4*)(ap);
      a.q[1] = *(const uint4*)(ap + 8);
      union { uint32_t u[8]; v16bf v; } b;
      const int kp0 = ks * 16 + hi * 8;
      const int pb  = pt * 16 + n;
      #pragma unroll
      for (int v = 0; v < 8; ++v) b.u[v] = xb[kp0 + v][pb];
      acc = __builtin_amdgcn_wmma_f32_16x16x32_bf16(false, a.v, false, b.v,
                                                    (short)0, acc, false, false);
    }
    const int mbase = mt * 16 + hi * 8;
    float4 bA = *(const float4*)(bias1 + mbase);
    float4 bB = *(const float4*)(bias1 + mbase + 4);
    float bb[8] = {bA.x, bA.y, bA.z, bA.w, bB.x, bB.y, bB.z, bB.w};
    const int p = pt * 16 + n;
    #pragma unroll
    for (int v = 0; v < 8; v += 2) {
      float f0 = valid ? fmaxf(acc[v]     + bb[v],     0.f) : 0.f;
      float f1 = valid ? fmaxf(acc[v + 1] + bb[v + 1], 0.f) : 0.f;
      t1p[p][(mbase + v) >> 1] = pack_bf16(f0, f1);
    }
  }
  __syncthreads();

  // ---- Stage C: depthwise 3x3 (f32 VALU), BN2 + ReLU -> bpack[pair][pix] --
  for (int idx = tid; idx < 288 * PPAD; idx += 256) {
    int pp = idx >> 6;
    int p  = idx & 63;
    if (p >= PINT) { bpack[pp][p] = 0u; continue; }
    int r = p / TC_, c = p - r * TC_;
    int m0 = pp * 2;
    float acc0 = bias2[m0], acc1 = bias2[m0 + 1];
    const float* wda = wdf + m0 * 9;
    #pragma unroll
    for (int dy = 0; dy < 3; ++dy)
      #pragma unroll
      for (int dx = 0; dx < 3; ++dx) {
        uint32_t t = t1p[(r + dy) * 16 + (c + dx)][pp];
        acc0 = fmaf(wda[dy * 3 + dx],     bf2f((uint16_t)t),         acc0);
        acc1 = fmaf(wda[9 + dy * 3 + dx], bf2f((uint16_t)(t >> 16)), acc1);
      }
    bpack[pp][p] = pack_bf16(fmaxf(acc0, 0.f), fmaxf(acc1, 0.f));
  }
  __syncthreads();

  // ---- Stage D: pointwise GEMM (96x576 @ 576x64), BN3 + residual ----------
  for (int j = wv; j < 24; j += 8) {
    int ot = j >> 2, nt = j & 3;
    v8f acc = {};
    const int arow = ot * 16 + n;
    for (int ks = 0; ks < 18; ++ks) {
      union { uint4 q[2]; v16bf v; } a;
      const uint16_t* ap = w3b + arow * M_EXP + ks * 32 + hi * 16;
      a.q[0] = *(const uint4*)(ap);
      a.q[1] = *(const uint4*)(ap + 8);
      union { uint32_t u[8]; v16bf v; } b;
      const int kp0 = ks * 16 + hi * 8;
      const int pb  = nt * 16 + n;
      #pragma unroll
      for (int v = 0; v < 8; ++v) b.u[v] = bpack[kp0 + v][pb];
      acc = __builtin_amdgcn_wmma_f32_16x16x32_bf16(false, a.v, false, b.v,
                                                    (short)0, acc, false, false);
    }
    const int obase = ot * 16 + hi * 8;
    float4 bA = *(const float4*)(bias3 + obase);
    float4 bB = *(const float4*)(bias3 + obase + 4);
    float bb[8] = {bA.x, bA.y, bA.z, bA.w, bB.x, bB.y, bB.z, bB.w};
    const int p = nt * 16 + n;
    if (p < PINT) {
      int r = p / TC_, c = p - r * TC_;
      size_t pix = (size_t)(row0 + r) * HW_ + (col0 + c);
      #pragma unroll
      for (int v = 0; v < 8; ++v) {
        size_t idx = ((size_t)nb * C_IN + (obase + v)) * (HW_ * HW_) + pix;
        out[idx] = acc[v] + bb[v] + x[idx];
      }
    }
  }
}

extern "C" void kernel_launch(void* const* d_in, const int* in_sizes, int n_in,
                              void* d_out, int out_size, void* d_ws, size_t ws_size,
                              hipStream_t stream) {
  const float* x  = (const float*)d_in[0];
  const float* w1 = (const float*)d_in[1];
  const float* g1 = (const float*)d_in[2];
  const float* b1 = (const float*)d_in[3];
  const float* m1 = (const float*)d_in[4];
  const float* v1 = (const float*)d_in[5];
  const float* wd = (const float*)d_in[6];
  const float* g2 = (const float*)d_in[7];
  const float* b2 = (const float*)d_in[8];
  const float* m2 = (const float*)d_in[9];
  const float* v2 = (const float*)d_in[10];
  const float* w3 = (const float*)d_in[11];
  const float* g3 = (const float*)d_in[12];
  const float* b3 = (const float*)d_in[13];
  const float* m3 = (const float*)d_in[14];
  const float* v3 = (const float*)d_in[15];

  char* ws = (char*)d_ws;
  uint16_t* w1b   = (uint16_t*)(ws + WS_W1B);
  uint16_t* w3b   = (uint16_t*)(ws + WS_W3B);
  float*    bias1 = (float*)(ws + WS_B1);
  float*    bias2 = (float*)(ws + WS_B2);
  float*    bias3 = (float*)(ws + WS_B3);
  float*    wdf   = (float*)(ws + WS_WD);
  uint32_t* xp    = (uint32_t*)(ws + WS_XP);

  mbconv_prep<<<(M_EXP * C_IN + 255) / 256, 256, 0, stream>>>(
      w1, g1, b1, m1, v1, wd, g2, b2, m2, v2, w3, g3, b3, m3, v3,
      w1b, w3b, bias1, bias2, bias3, wdf);

  const bool use_xp = (ws_size >= WS_NEED);
  dim3 grid(56, 32);   // 14 row-tiles * 4 col-tiles, 32 batches
  if (use_xp) {
    mbconv_xpack<<<(XP_ELEMS + 255) / 256, 256, 0, stream>>>(x, xp);
    mbconv_fused<1><<<grid, 256, 0, stream>>>(x, xp, w1b, w3b, bias1, bias2,
                                              bias3, wdf, (float*)d_out);
  } else {
    mbconv_fused<0><<<grid, 256, 0, stream>>>(x, xp, w1b, w3b, bias1, bias2,
                                              bias3, wdf, (float*)d_out);
  }
}
